// LiquidMoERouter_65841848648374
// MI455X (gfx1250) — matrix operand order repaired and checked by reading, not compiled, not run on hardware
//
#include <hip/hip_runtime.h>
#include <hip/hip_bf16.h>
#include <math.h>

// Problem constants (from reference)
#define N_TOK  8192
#define IN_DIM 4096
#define HID    1024
#define NEXP   64
#define DT_C   0.02f

// GEMM1 blocking
#define BK      32                 // K per WMMA step
#define BROWS   64                 // B-panel rows (N columns) per block
#define PITCH   80                 // 64B row + 16B TDM pad -> 16B aligned, conflict-free
#define PANEL   (BROWS * PITCH)    // 5120 B per buffer
#define KSTEPS  (IN_DIM / BK)      // 128

typedef __attribute__((ext_vector_type(16))) __bf16 v16bf;
typedef __attribute__((ext_vector_type(8)))  __bf16 v8bf;
typedef __attribute__((ext_vector_type(8)))  float  v8f;
typedef __attribute__((ext_vector_type(4)))  unsigned int u32x4;
typedef __attribute__((ext_vector_type(8)))  int i32x8;
typedef __attribute__((ext_vector_type(4)))  int i32x4;

#if __has_builtin(__builtin_amdgcn_tensor_load_to_lds)
#define USE_TDM 1
#else
#define USE_TDM 0
#endif

// ---------------------------------------------------------------------------
// fp32 -> bf16 cast (U_w, G_w: heavily reused, convert once)
// ---------------------------------------------------------------------------
__global__ void cast_f32_bf16(const float* __restrict__ src,
                              __bf16* __restrict__ dst, int n) {
  int i = blockIdx.x * blockDim.x + threadIdx.x;
  if (i < n) dst[i] = (__bf16)src[i];
}

// ---------------------------------------------------------------------------
// GEMM1 + liquid-cell epilogue: h = 0.02*tanh(x @ U_w^T + U_b + W_b)
// Block = 256 threads (8 waves) computes a 128(M) x 64(N) output block.
// B-panel (64 rows x 32 K, bf16) is staged to LDS by the Tensor Data Mover
// (double-buffered, TENSORcnt + barrier), with TDM LDS padding for
// bank-conflict-free ds_load_b128 fragment reads. Each wave owns one 16-row
// strip and 4 N-tiles: 1 A fragment feeds 4 WMMAs per K step.
// ---------------------------------------------------------------------------
__global__ __launch_bounds__(256) void gemm1_tanh(
    const float*  __restrict__ x,    // [N_TOK][IN_DIM] fp32
    const __bf16* __restrict__ ub,   // [HID][IN_DIM]   bf16
    const float*  __restrict__ U_b,  // [HID]
    const float*  __restrict__ W_b,  // [HID]
    __bf16*       __restrict__ hb)   // [N_TOK][HID]    bf16 out
{
  __shared__ __align__(32) char smem[2 * PANEL];

  const int wave = threadIdx.x >> 5;
  const int lane = threadIdx.x & 31;
  const int g = lane >> 4;                 // lane-half -> K group
  const int r = lane & 15;                 // A row / B col within tile
  const int m0 = (blockIdx.x >> 4) * 128;  // 64 M blocks
  const int n0 = (blockIdx.x & 15) * BROWS;// 16 N blocks

  const float* arow = x + (size_t)(m0 + wave * 16 + r) * IN_DIM + g * 8;

#if USE_TDM
  const unsigned lds0 = (unsigned)(size_t)(void*)smem;   // LDS byte offset
  const unsigned long long gbase = (unsigned long long)(const void*)ub;
#endif

  // Stage B-panel for K-step `ks` into LDS buffer `buf`.
  auto stage = [&](int ks, int buf) {
#if USE_TDM
    if (wave == 0) {
      const unsigned long long ga =
          gbase + ((unsigned long long)n0 * IN_DIM + (unsigned)(ks * BK)) * 2ull;
      u32x4 g0;
      g0[0] = 1u;                                  // count=1, user desc
      g0[1] = lds0 + (unsigned)buf * PANEL;        // lds_addr (bytes)
      g0[2] = (unsigned)(ga & 0xFFFFFFFFull);      // global_addr[31:0]
      g0[3] = (unsigned)((ga >> 32) & 0x01FFFFFFull) | 0x80000000u; // type=2
      i32x8 g1;
      // data_size=1 (2B) | pad_enable | pad_interval=3 (64B) | pad_amount=3 (16B)
      g1[0] = (int)((1u << 16) | (1u << 20) | (3u << 22) | (3u << 25));
      g1[1] = (int)(((unsigned)IN_DIM & 0xFFFFu) << 16); // tensor_dim0[15:0]
      g1[2] = (int)(((unsigned)HID & 0xFFFFu) << 16);    // dim0 hi | tensor_dim1[15:0]
      g1[3] = (int)((unsigned)BK << 16);                 // dim1 hi | tile_dim0=32
      g1[4] = BROWS;                                     // tile_dim1=64 | tile_dim2=0
      g1[5] = IN_DIM;                                    // tensor_dim0_stride lo32
      g1[6] = 0;                                         // stride0 hi | stride1 lo
      g1[7] = 0;                                         // stride1 hi
      i32x4 z4 = {0, 0, 0, 0};
#if __clang_major__ >= 23
      i32x8 z8 = {0, 0, 0, 0, 0, 0, 0, 0};
      __builtin_amdgcn_tensor_load_to_lds(g0, g1, z4, z4, z8, 0);
#else
      __builtin_amdgcn_tensor_load_to_lds(g0, g1, z4, z4, 0);
#endif
    }
#else
    // Cooperative copy fallback: 64 rows x 4 chunks of 16B, 256 threads.
    const int row = threadIdx.x >> 2;
    const int c   = threadIdx.x & 3;
    const __bf16* src = ub + (size_t)(n0 + row) * IN_DIM + ks * BK + c * 8;
    *(v8bf*)(smem + buf * PANEL + row * PITCH + c * 16) = *(const v8bf*)src;
#endif
  };
  auto stage_wait = [&]() {
#if USE_TDM
    if (wave == 0) __builtin_amdgcn_s_wait_tensorcnt(0);
#endif
  };

  v8f acc[4] = {};

  stage(0, 0);
  stage_wait();
  __syncthreads();

  for (int ks = 0; ks < KSTEPS; ++ks) {
    const int cur = ks & 1;
    if (ks + 1 < KSTEPS) stage(ks + 1, cur ^ 1);   // DMA next panel (overlaps)

    const int k = ks * BK;
    __builtin_prefetch(arow + k + 1024, 0, 0);     // global_prefetch_b8

    // A fragment: fp32 -> bf16 on the fly (ISA 16-bit A layout: K chunks
    // g*8..+7 and 16+g*8..+7)
    v8f a0 = *(const v8f*)(arow + k);
    v8f a1 = *(const v8f*)(arow + k + 16);
    v16bf afrag;
    #pragma unroll
    for (int i = 0; i < 8; ++i) {
      afrag[i]     = (__bf16)a0[i];
      afrag[8 + i] = (__bf16)a1[i];
    }

    const char* bb = smem + cur * PANEL + g * 16;
    #pragma unroll
    for (int t = 0; t < 4; ++t) {
      const char* p = bb + (t * 16 + r) * PITCH;
      v8bf blo = *(const v8bf*)p;                  // ds_load_b128
      v8bf bhi = *(const v8bf*)(p + 32);           // ds_load_b128
      v16bf bfrag = __builtin_shufflevector(blo, bhi,
          0, 1, 2, 3, 4, 5, 6, 7, 8, 9, 10, 11, 12, 13, 14, 15);
      acc[t] = __builtin_amdgcn_wmma_f32_16x16x32_bf16(
          false, afrag, false, bfrag, (short)0, acc[t], false, false);
    }

    if (ks + 1 < KSTEPS) stage_wait();             // s_wait_tensorcnt 0
    __syncthreads();
  }

  // Epilogue: C/D layout -> lane holds N=r, VGPR i holds M = g*8 + i
  #pragma unroll
  for (int t = 0; t < 4; ++t) {
    const int   col  = n0 + t * 16 + r;
    const float bias = U_b[col] + W_b[col];        // h_prev==0: only biases
    #pragma unroll
    for (int i = 0; i < 8; ++i) {
      const int row = m0 + wave * 16 + g * 8 + i;
      hb[(size_t)row * HID + col] = (__bf16)(DT_C * tanhf(acc[t][i] + bias));
    }
  }
}

// ---------------------------------------------------------------------------
// GEMM2: logits[m][e] = h @ G_w^T + G_b. One wave per 16-row strip computes
// all 4 expert tiles (A fragment reused 4x). G_w panel is L2/LDS-trivial.
// ---------------------------------------------------------------------------
__global__ __launch_bounds__(256) void gemm2_logits(
    const __bf16* __restrict__ hb,     // [N_TOK][HID] bf16
    const __bf16* __restrict__ gb,     // [NEXP][HID]  bf16
    const float*  __restrict__ G_b,    // [NEXP]
    float*        __restrict__ logits) // [N_TOK][NEXP]
{
  const int wave = threadIdx.x >> 5;
  const int lane = threadIdx.x & 31;
  const int g = lane >> 4;
  const int r = lane & 15;
  const int m0 = (blockIdx.x * 8 + wave) * 16;     // 512 strips

  const __bf16* arow = hb + (size_t)(m0 + r) * HID + g * 16;

  v8f acc[4] = {};
  #pragma unroll 2
  for (int k = 0; k < HID; k += 32) {
    v8bf alo = *(const v8bf*)(arow + k);
    v8bf ahi = *(const v8bf*)(arow + k + 8);
    v16bf afrag = __builtin_shufflevector(alo, ahi,
        0, 1, 2, 3, 4, 5, 6, 7, 8, 9, 10, 11, 12, 13, 14, 15);
    #pragma unroll
    for (int t = 0; t < 4; ++t) {
      v16bf bfrag = *(const v16bf*)(gb + (size_t)(t * 16 + r) * HID + g * 16 + k);
      acc[t] = __builtin_amdgcn_wmma_f32_16x16x32_bf16(
          false, afrag, false, bfrag, (short)0, acc[t], false, false);
    }
  }

  #pragma unroll
  for (int t = 0; t < 4; ++t) {
    const int   col  = t * 16 + r;
    const float bias = G_b[col];
    #pragma unroll
    for (int i = 0; i < 8; ++i)
      logits[(size_t)(m0 + g * 8 + i) * NEXP + col] = acc[t][i] + bias;
  }
}

// ---------------------------------------------------------------------------
// Per-token softmax over 64 experts, top-2 (first-occurrence tie-break, like
// lax.top_k), renormalized. Output: [w 8192*2 | idx 8192*2 (float) | p 8192*64]
// ---------------------------------------------------------------------------
__global__ void softmax_top2(const float* __restrict__ logits,
                             float* __restrict__ out) {
  const int n = blockIdx.x * blockDim.x + threadIdx.x;
  if (n >= N_TOK) return;
  const float* l = logits + (size_t)n * NEXP;

  float mx = -INFINITY;
  #pragma unroll
  for (int j = 0; j < NEXP; ++j) mx = fmaxf(mx, l[j]);

  float e[NEXP];
  float sum = 0.f;
  #pragma unroll
  for (int j = 0; j < NEXP; ++j) { e[j] = __expf(l[j] - mx); sum += e[j]; }
  const float inv = 1.0f / sum;

  float* probs = out + 2 * N_TOK * 2;
  float b1 = -1.f, b2 = -1.f;
  int   i1 = 0,   i2 = 0;
  #pragma unroll
  for (int j = 0; j < NEXP; ++j) {
    const float p = e[j] * inv;
    probs[(size_t)n * NEXP + j] = p;
    if (p > b1)      { b2 = b1; i2 = i1; b1 = p; i1 = j; }
    else if (p > b2) { b2 = p;  i2 = j; }
  }
  const float denom = b1 + b2 + 1e-8f;
  out[n * 2 + 0] = b1 / denom;
  out[n * 2 + 1] = b2 / denom;
  out[N_TOK * 2 + n * 2 + 0] = (float)i1;
  out[N_TOK * 2 + n * 2 + 1] = (float)i2;
}

// ---------------------------------------------------------------------------
extern "C" void kernel_launch(void* const* d_in, const int* in_sizes, int n_in,
                              void* d_out, int out_size, void* d_ws, size_t ws_size,
                              hipStream_t stream) {
  // Input order: x, W_w, W_b, U_w, U_b, V_w, V_b, G_w, G_b
  const float* x   = (const float*)d_in[0];
  const float* W_b = (const float*)d_in[2];
  const float* U_w = (const float*)d_in[3];
  const float* U_b = (const float*)d_in[4];
  const float* G_w = (const float*)d_in[7];
  const float* G_b = (const float*)d_in[8];
  // W_w multiplies h_prev==0; V_w/V_b feed only the dead tau path.

  char* ws = (char*)d_ws;
  __bf16* ub     = (__bf16*)(ws);                    //  8 MiB  [HID][IN_DIM]
  __bf16* gb     = (__bf16*)(ws + 8388608);          //  128KiB [NEXP][HID]
  __bf16* hb     = (__bf16*)(ws + 8519680);          // 16 MiB  [N_TOK][HID]
  float*  logits = (float*) (ws + 25296896);         //  2 MiB  [N_TOK][NEXP]
  float*  out    = (float*)d_out;

  cast_f32_bf16<<<(HID * IN_DIM + 255) / 256, 256, 0, stream>>>(U_w, ub, HID * IN_DIM);
  cast_f32_bf16<<<(NEXP * HID + 255) / 256, 256, 0, stream>>>(G_w, gb, NEXP * HID);

  // 64 M-blocks x 16 N-blocks, 8 waves each
  gemm1_tanh<<<1024, 256, 0, stream>>>(x, ub, U_b, W_b, hb);
  // 512 strips / 8 waves
  gemm2_logits<<<64, 256, 0, stream>>>(hb, gb, G_b, logits);

  softmax_top2<<<(N_TOK + 255) / 256, 256, 0, stream>>>(logits, out);
}
